// DecoderBlock_28467043238485
// MI455X (gfx1250) — matrix operand order repaired and checked by reading, not compiled, not run on hardware
//
#include <hip/hip_runtime.h>
#include <cstdint>

// Problem constants (fixed by the reference).
#define B_    2
#define S_    2048
#define D_    512
#define H_    8
#define DH_   64
#define DFF_  2048
#define MROWS (B_ * S_)   // 4096 token rows

// ---------------------------------------------------------------------------
// Types for CDNA5 WMMA (wave32): v16bf A/B fragments, v8f accumulator.
// ---------------------------------------------------------------------------
typedef __bf16 v16bf __attribute__((ext_vector_type(16)));
typedef __bf16 v8bf  __attribute__((ext_vector_type(8)));
typedef __bf16 v4bf  __attribute__((ext_vector_type(4)));
typedef float  v8f   __attribute__((ext_vector_type(8)));
typedef float  v4f   __attribute__((ext_vector_type(4)));

// Native converts -> v_cvt_*bf16* hardware ops (gfx1250 has bf16 units).
__device__ __forceinline__ v4bf cvt4(v4f x) {
  v4bf r;
  r[0] = (__bf16)x[0]; r[1] = (__bf16)x[1];
  r[2] = (__bf16)x[2]; r[3] = (__bf16)x[3];
  return r;
}

union FragU { v16bf v; v8bf h[2]; };

// A fragment (16x32 bf16): lanes 0-15 hold row M=lane, K chunks {0-7,16-23};
// lanes 16-31 hold row M=lane-16, K chunks {8-15,24-31}.  Two 16B LDS loads.
__device__ __forceinline__ v16bf ld_frag_a(const __bf16* base, int stride,
                                           int m0, int k0, int lane) {
  const __bf16* p = base + (size_t)(m0 + (lane & 15)) * stride + k0 + ((lane & 16) ? 8 : 0);
  FragU f;
  f.h[0] = *(const v8bf*)(p);
  f.h[1] = *(const v8bf*)(p + 16);
  return f.v;
}

// B fragment (32x16 bf16) from a [N][K] (K-contiguous) LDS tile:
// lanes 0-15 hold col N=lane with K chunks {0-7,16-23}; lanes 16-31 hold K chunks {8-15,24-31}.
__device__ __forceinline__ v16bf ld_frag_b(const __bf16* base, int stride,
                                           int n0, int k0, int lane) {
  const __bf16* p = base + (size_t)(n0 + (lane & 15)) * stride + k0 + ((lane & 16) ? 8 : 0);
  FragU f;
  f.h[0] = *(const v8bf*)(p);
  f.h[1] = *(const v8bf*)(p + 16);
  return f.v;
}

__device__ __forceinline__ v8f wmma_bf16(v16bf a, v16bf b, v8f c) {
  // 8-arg form: (neg_a, A, neg_b, B, c_mod, C, reuse_a, reuse_b)
  return __builtin_amdgcn_wmma_f32_16x16x32_bf16(false, a, false, b,
                                                 (short)0, c, false, false);
}

// ---------------------------------------------------------------------------
// GEMM: C[M,N] = A[M,K] @ W[N,K]^T + bias[N]   (torch Linear semantics)
// Block tile 128x128, K-step 64. 8 waves: 4(M) x 2(N), each wave 32x64.
// 16 WMMAs per wave per K-step.  BHSD: scatter output as [B,H,S,DH].
// ---------------------------------------------------------------------------
template<bool RELU, bool BHSD>
__global__ __launch_bounds__(256) void gemm_wmma(const float* __restrict__ A,
                                                 const float* __restrict__ W,
                                                 const float* __restrict__ bias,
                                                 float* __restrict__ C,
                                                 int M, int N, int K) {
  __shared__ __attribute__((aligned(16))) __bf16 As[128 * 72];
  __shared__ __attribute__((aligned(16))) __bf16 Bs[128 * 72];
  const int tid  = threadIdx.x;
  const int lane = tid & 31, wave = tid >> 5;
  const int wm = (wave >> 1) * 32, wn = (wave & 1) * 64;
  const int n0 = blockIdx.x * 128, m0 = blockIdx.y * 128;

  v8f zero = {};
  v8f acc[2][4];
#pragma unroll
  for (int i = 0; i < 2; ++i)
#pragma unroll
    for (int j = 0; j < 4; ++j) acc[i][j] = zero;

  for (int k0 = 0; k0 < K; k0 += 64) {
    __syncthreads();
    // Stage A and W tiles (f32 -> bf16, packed 4-wide), 8 float4 rows/thread.
    for (int i = tid; i < 2048; i += 256) {
      int r = i >> 4, c4 = (i & 15) * 4;
      const float* ap = A + (size_t)(m0 + r) * K + k0 + c4;
      const float* wp = W + (size_t)(n0 + r) * K + k0 + c4;
      *(v4bf*)(As + r * 72 + c4) = cvt4(*(const v4f*)ap);
      *(v4bf*)(Bs + r * 72 + c4) = cvt4(*(const v4f*)wp);
      if (k0 + 64 < K) {               // prefetch next K tile into L2/L0
        __builtin_prefetch(ap + 64, 0, 0);
        __builtin_prefetch(wp + 64, 0, 0);
      }
    }
    __syncthreads();

#pragma unroll
    for (int kc = 0; kc < 2; ++kc) {
      v16bf a0 = ld_frag_a(As, 72, wm + 0,  kc * 32, lane);
      v16bf a1 = ld_frag_a(As, 72, wm + 16, kc * 32, lane);
#pragma unroll
      for (int fn = 0; fn < 4; ++fn) {
        v16bf bfr = ld_frag_b(Bs, 72, wn + fn * 16, kc * 32, lane);
        acc[0][fn] = wmma_bf16(a0, bfr, acc[0][fn]);
        acc[1][fn] = wmma_bf16(a1, bfr, acc[1][fn]);
      }
    }
  }

  // Epilogue. C-matrix layout: VGPR r holds M=r (lanes 0-15) / M=r+8 (16-31),
  // N = lane%16.
  const int hi = (lane >> 4) << 3;
  const int ln = lane & 15;
#pragma unroll
  for (int fm = 0; fm < 2; ++fm)
#pragma unroll
    for (int fn = 0; fn < 4; ++fn) {
      int n = n0 + wn + fn * 16 + ln;
      float bn = bias[n];
#pragma unroll
      for (int r = 0; r < 8; ++r) {
        int m = m0 + wm + fm * 16 + r + hi;
        float val = acc[fm][fn][r] + bn;
        if (RELU) val = fmaxf(val, 0.0f);
        if (BHSD) {
          int bb = m >> 11, ss = m & (S_ - 1);
          int hh = n >> 6,  dd = n & (DH_ - 1);
          C[(((size_t)bb * H_ + hh) * S_ + ss) * DH_ + dd] = val;
        } else {
          C[(size_t)m * N + n] = val;
        }
      }
    }
}

// ---------------------------------------------------------------------------
// Flash-style causal attention.  One workgroup = 64 query rows of one (b,h);
// 4 waves, each owning 16 rows.  Key blocks of 128, online softmax.
// ---------------------------------------------------------------------------
__global__ __launch_bounds__(128) void attn_flash(const float* __restrict__ Qg,
                                                  const float* __restrict__ Kg,
                                                  const float* __restrict__ Vg,
                                                  const float* __restrict__ alphas,
                                                  float* __restrict__ ctx) {
  __shared__ __attribute__((aligned(16))) __bf16 Qs[64 * 72];        // [q][d]
  __shared__ __attribute__((aligned(16))) __bf16 Ks[128 * 72];       // [k][d]
  __shared__ __attribute__((aligned(16))) __bf16 Vt[64 * 136];       // [d][k] (transposed)
  __shared__ __attribute__((aligned(16))) __bf16 Ps[4 * 16 * 136];   // per-wave P tile
  const int tid = threadIdx.x, lane = tid & 31, w = tid >> 5;
  const int qb = blockIdx.x, bh = blockIdx.y;
  const int b = bh >> 3, h = bh & 7;
  const float* qp = Qg + (size_t)bh * S_ * DH_ + (size_t)qb * 64 * DH_;
  const float* kp = Kg + (size_t)bh * S_ * DH_;
  const float* vp = Vg + (size_t)bh * S_ * DH_;

  // Stage Q (64x64 f32 -> bf16, packed).
  for (int i = tid; i < 64 * 16; i += 128) {
    int r = i >> 4, c4 = (i & 15) * 4;
    *(v4bf*)(Qs + r * 72 + c4) = cvt4(*(const v4f*)(qp + r * DH_ + c4));
  }

  const int hi = (lane >> 4) << 3;
  const int ln = lane & 15;
  const int qbase = qb * 64 + w * 16;

  float mrow[8], lrow[8];
  v8f zero = {};
  v8f o[4];
#pragma unroll
  for (int r = 0; r < 8; ++r) { mrow[r] = -1e30f; lrow[r] = 0.0f; }
#pragma unroll
  for (int i = 0; i < 4; ++i) o[i] = zero;

  const int nkb = (qb >> 1) + 1;   // causal: key blocks 0 .. qb/2
  for (int kb = 0; kb < nkb; ++kb) {
    __syncthreads();
    // Stage K tile [128][64] and V transposed [64][128].
    for (int i = tid; i < 128 * 16; i += 128) {
      int r = i >> 4, c4 = (i & 15) * 4;
      *(v4bf*)(Ks + r * 72 + c4) =
          cvt4(*(const v4f*)(kp + (size_t)(kb * 128 + r) * DH_ + c4));
      v4f tv = *(const v4f*)(vp + (size_t)(kb * 128 + r) * DH_ + c4);
      Vt[(c4 + 0) * 136 + r] = (__bf16)tv[0];
      Vt[(c4 + 1) * 136 + r] = (__bf16)tv[1];
      Vt[(c4 + 2) * 136 + r] = (__bf16)tv[2];
      Vt[(c4 + 3) * 136 + r] = (__bf16)tv[3];
    }
    __syncthreads();

    // S = Q K^T (16 x 128 per wave): 8 N-fragments, d reduced in 2 chunks.
    v16bf qa0 = ld_frag_a(Qs, 72, w * 16, 0,  lane);
    v16bf qa1 = ld_frag_a(Qs, 72, w * 16, 32, lane);
    v8f s[8];
#pragma unroll
    for (int fn = 0; fn < 8; ++fn) {
      v16bf kb0 = ld_frag_b(Ks, 72, fn * 16, 0,  lane);
      v16bf kb1 = ld_frag_b(Ks, 72, fn * 16, 32, lane);
      v8f a = zero;
      a = wmma_bf16(qa0, kb0, a);
      a = wmma_bf16(qa1, kb1, a);
      s[fn] = a;
    }

    // Scale, causal mask, row max.
    float mx[8];
#pragma unroll
    for (int r = 0; r < 8; ++r) mx[r] = -1e30f;
#pragma unroll
    for (int fn = 0; fn < 8; ++fn) {
      int kg = kb * 128 + fn * 16 + ln;
#pragma unroll
      for (int r = 0; r < 8; ++r) {
        float vv = s[fn][r] * 0.125f;            // 1/sqrt(64)
        if (kg > qbase + r + hi) vv = -1e30f;    // causal
        s[fn][r] = vv;
        mx[r] = fmaxf(mx[r], vv);
      }
    }
    // Row m lives entirely within one 16-lane half: xor-reduce over 16 lanes.
#pragma unroll
    for (int r = 0; r < 8; ++r)
#pragma unroll
      for (int off = 1; off < 16; off <<= 1)
        mx[r] = fmaxf(mx[r], __shfl_xor(mx[r], off, 32));

    float corr[8], rs[8];
#pragma unroll
    for (int r = 0; r < 8; ++r) {
      float mn = fmaxf(mrow[r], mx[r]);
      corr[r] = __expf(mrow[r] - mn);
      mrow[r] = mn;
      rs[r] = 0.0f;
    }
#pragma unroll
    for (int fn = 0; fn < 8; ++fn)
#pragma unroll
      for (int r = 0; r < 8; ++r) {
        float p = __expf(s[fn][r] - mrow[r]);
        s[fn][r] = p;
        rs[r] += p;
      }
#pragma unroll
    for (int r = 0; r < 8; ++r) {
#pragma unroll
      for (int off = 1; off < 16; off <<= 1)
        rs[r] += __shfl_xor(rs[r], off, 32);
      lrow[r] = lrow[r] * corr[r] + rs[r];
    }
#pragma unroll
    for (int fd = 0; fd < 4; ++fd)
#pragma unroll
      for (int r = 0; r < 8; ++r) o[fd][r] *= corr[r];

    // Dump P (16x128) to this wave's LDS region as bf16 (A-fragment source).
    __bf16* Pw = Ps + w * 16 * 136;
#pragma unroll
    for (int fn = 0; fn < 8; ++fn)
#pragma unroll
      for (int r = 0; r < 8; ++r)
        Pw[(r + hi) * 136 + fn * 16 + ln] = (__bf16)s[fn][r];

    // O += P V  (reduce over 128 keys in 4 chunks of 32; 4 d-fragments).
#pragma unroll
    for (int kc = 0; kc < 4; ++kc) {
      v16bf pa = ld_frag_a(Pw, 136, 0, kc * 32, lane);
#pragma unroll
      for (int fd = 0; fd < 4; ++fd) {
        v16bf vb = ld_frag_b(Vt, 136, fd * 16, kc * 32, lane);
        o[fd] = wmma_bf16(pa, vb, o[fd]);
      }
    }
  }

  // Normalize, scale by head alpha, store ctx in [B,S,D] layout.
  const float alpha = alphas[h];
#pragma unroll
  for (int fd = 0; fd < 4; ++fd)
#pragma unroll
    for (int r = 0; r < 8; ++r) {
      int srow = qbase + r + hi;
      int d = h * DH_ + fd * 16 + ln;
      ctx[((size_t)b * S_ + srow) * D_ + d] = o[fd][r] / lrow[r] * alpha;
    }
}

// ---------------------------------------------------------------------------
// out = LayerNorm(A + B) * g + beta, one 256-thread block per row (D=512).
// ---------------------------------------------------------------------------
__global__ __launch_bounds__(256) void add_ln(const float* __restrict__ A,
                                              const float* __restrict__ Bv,
                                              const float* __restrict__ g,
                                              const float* __restrict__ be,
                                              float* __restrict__ out) {
  __shared__ float red[256];
  const int row = blockIdx.x, tid = threadIdx.x;
  const float* a = A  + (size_t)row * D_;
  const float* b = Bv + (size_t)row * D_;
  float x0 = a[tid] + b[tid];
  float x1 = a[tid + 256] + b[tid + 256];

  red[tid] = x0 + x1;
  __syncthreads();
  for (int st = 128; st > 0; st >>= 1) {
    if (tid < st) red[tid] += red[tid + st];
    __syncthreads();
  }
  float mean = red[0] * (1.0f / D_);
  __syncthreads();

  float d0 = x0 - mean, d1 = x1 - mean;
  red[tid] = d0 * d0 + d1 * d1;
  __syncthreads();
  for (int st = 128; st > 0; st >>= 1) {
    if (tid < st) red[tid] += red[tid + st];
    __syncthreads();
  }
  float inv = rsqrtf(red[0] * (1.0f / D_) + 1e-5f);
  out[(size_t)row * D_ + tid]       = d0 * inv * g[tid]       + be[tid];
  out[(size_t)row * D_ + tid + 256] = d1 * inv * g[tid + 256] + be[tid + 256];
}

// ---------------------------------------------------------------------------
// Launcher.  Input order follows setup_inputs(); attn_mask (d_in[1]) is the
// standard causal mask, implemented directly in attn_flash.
// Workspace layout (floats): q,k,v,ctx,attn_out,h1 (2M each), a1 (8M), ff (2M)
// -> ~92 MB total.
// ---------------------------------------------------------------------------
extern "C" void kernel_launch(void* const* d_in, const int* in_sizes, int n_in,
                              void* d_out, int out_size, void* d_ws, size_t ws_size,
                              hipStream_t stream) {
  (void)in_sizes; (void)n_in; (void)out_size; (void)ws_size;
  const float* x   = (const float*)d_in[0];
  const float* Wq  = (const float*)d_in[2];
  const float* bq  = (const float*)d_in[3];
  const float* Wk  = (const float*)d_in[4];
  const float* bk  = (const float*)d_in[5];
  const float* Wv  = (const float*)d_in[6];
  const float* bv  = (const float*)d_in[7];
  const float* Wo  = (const float*)d_in[8];
  const float* bo  = (const float*)d_in[9];
  const float* al  = (const float*)d_in[10];
  const float* g1  = (const float*)d_in[11];
  const float* be1 = (const float*)d_in[12];
  const float* W1  = (const float*)d_in[13];
  const float* b1  = (const float*)d_in[14];
  const float* W2  = (const float*)d_in[15];
  const float* b2  = (const float*)d_in[16];
  const float* g2  = (const float*)d_in[17];
  const float* be2 = (const float*)d_in[18];

  float* ws = (float*)d_ws;
  const size_t SZ = (size_t)MROWS * D_;   // 2,097,152 floats
  float* q   = ws;
  float* k   = q   + SZ;
  float* v   = k   + SZ;
  float* ctx = v   + SZ;
  float* ao  = ctx + SZ;
  float* h1  = ao  + SZ;
  float* a1  = h1  + SZ;                  // MROWS * DFF floats
  float* ff  = a1  + (size_t)MROWS * DFF_;

  dim3 blk(256);
  dim3 gD(D_ / 128,  MROWS / 128);    //  4 x 32
  dim3 gF(DFF_ / 128, MROWS / 128);   // 16 x 32

  // Q,K,V projections -> [B,H,S,DH]
  gemm_wmma<false, true ><<<gD, blk, 0, stream>>>(x, Wq, bq, q, MROWS, D_, D_);
  gemm_wmma<false, true ><<<gD, blk, 0, stream>>>(x, Wk, bk, k, MROWS, D_, D_);
  gemm_wmma<false, true ><<<gD, blk, 0, stream>>>(x, Wv, bv, v, MROWS, D_, D_);

  // Causal attention -> ctx [B,S,D]
  attn_flash<<<dim3(S_ / 64, B_ * H_), dim3(128), 0, stream>>>(q, k, v, al, ctx);

  // Output projection, residual + LN1
  gemm_wmma<false, false><<<gD, blk, 0, stream>>>(ctx, Wo, bo, ao, MROWS, D_, D_);
  add_ln<<<MROWS, blk, 0, stream>>>(x, ao, g1, be1, h1);

  // FFN: W1 + ReLU, W2, residual + LN2
  gemm_wmma<true,  false><<<gF, blk, 0, stream>>>(h1, W1, b1, a1, MROWS, DFF_, D_);
  gemm_wmma<false, false><<<gD, blk, 0, stream>>>(a1, W2, b2, ff, MROWS, D_, DFF_);
  add_ln<<<MROWS, blk, 0, stream>>>(h1, ff, g2, be2, (float*)d_out);
}
